// CatNet_training_80229989089858
// MI455X (gfx1250) — matrix operand order repaired and checked by reading, not compiled, not run on hardware
//
#include <hip/hip_runtime.h>
#include <hip/hip_bf16.h>

typedef __attribute__((ext_vector_type(16))) _Float16 v16h;
typedef __attribute__((ext_vector_type(8)))  _Float16 v8h;
typedef __attribute__((ext_vector_type(8)))  float    v8f;
typedef __attribute__((ext_vector_type(4)))  unsigned int u32x4;
typedef __attribute__((ext_vector_type(8)))  int      i32x8;
typedef __attribute__((ext_vector_type(4)))  int      i32x4;

#if defined(__has_builtin)
#if __has_builtin(__builtin_amdgcn_tensor_load_to_lds)
#define HAVE_TDM 1
#endif
#endif
#ifndef HAVE_TDM
#define HAVE_TDM 0
#endif

#if HAVE_TDM
// Tensor-DMA descriptor (D#) for a 2D tile: `rows` rows x `len` f16 elements,
// row stride = `len` (fully contiguous panel). Fields per CDNA5 ISA 8.3/8.4.
// 6-arg builtin: (g0 u32x4, g1 i32x8, g2 i32x4, g3 i32x4, g2b i32x8, cpol).
__device__ __forceinline__ void tdm_load_w_tile(unsigned lds_off,
                                                const _Float16* gsrc,
                                                int len, int rows) {
  unsigned long long ga = (unsigned long long)(size_t)gsrc;
  u32x4 g0;
  g0[0] = 1u;                                            // count=1, user-mode
  g0[1] = lds_off;                                       // lds_addr (bytes)
  g0[2] = (unsigned)(ga & 0xFFFFFFFFull);                // global_addr[31:0]
  g0[3] = (unsigned)((ga >> 32) & 0x01FFFFFFull) | (2u << 30); // addr[56:32]|type=2
  unsigned uL = (unsigned)len, uR = (unsigned)rows;
  i32x8 g1;
  g1[0] = (int)(1u << 16);                               // data_size = 1 (2 bytes)
  g1[1] = (int)((uL & 0xFFFFu) << 16);                   // tensor_dim0 lo16
  g1[2] = (int)((uL >> 16) | ((uR & 0xFFFFu) << 16));    // dim0 hi16 | tensor_dim1 lo16
  g1[3] = (int)((uR >> 16) | ((uL & 0xFFFFu) << 16));    // dim1 hi16 | tile_dim0 = len
  g1[4] = (int)uR;                                       // tile_dim1 = rows, tile_dim2=0
  g1[5] = (int)uL;                                       // tensor_dim0_stride[31:0]
  g1[6] = 0;                                             // stride hi, dim1_stride lo
  g1[7] = 0;
  i32x4 z4 = {0, 0, 0, 0};                               // groups 2/3 unused (2D)
  i32x8 z8 = {0, 0, 0, 0, 0, 0, 0, 0};
  __builtin_amdgcn_tensor_load_to_lds(g0, g1, z4, z4, z8, 0);
}
#endif

// ---------------------------------------------------------------------------
// Implicit-GEMM 3x3 conv via WMMA f16->f32.
//   X  : [B, H, W, Ci]  f16 (NHWC, Ci multiple of 32)
//   Wt : [9, Co, Ci]    f16 (tap-major, Ci contiguous)
//   Y  : [B, Ho, Wo, Co] f32, Ho = pad ? H : H-2
// Block: 256 threads = 8 waves covering 128 co x 64 pixels. Per tap, one TDM
// load stages the contiguous [128co x Ci] weight panel into LDS; the K loop
// then runs barrier-free: each A fragment (from LDS) feeds 4 WMMAs over 4
// pixel tiles (B fragments are single v16h global loads thanks to NHWC).
// ---------------------------------------------------------------------------
__global__ __launch_bounds__(256)
void conv3x3_wmma_f16(const _Float16* __restrict__ X,
                      const _Float16* __restrict__ Wt,
                      const float* __restrict__ bias,
                      float* __restrict__ Y,
                      int B, int H, int W, int Ci, int Co, int pad) {
  extern __shared__ _Float16 sW[];               // [128][Ci]
  const int Ho = pad ? H : H - 2;
  const int Wo = pad ? W : W - 2;
  const long P  = (long)B * Ho * Wo;
  const long p0 = (long)blockIdx.x * 64;
  const int coBlock = blockIdx.y * 128;
  const int lane = threadIdx.x & 31;
  const int wave = threadIdx.x >> 5;
  const int col  = lane & 15;   // N (pixel) / M (co) index within tile
  const int kh   = lane >> 4;   // selects K half of the fragment

  // Pixel geometry for the wave's 4 tiles (p = p0 + 16*j + col).
  bool pv[4]; int bj[4], ohj[4], owj[4];
#pragma unroll
  for (int j = 0; j < 4; ++j) {
    long p = p0 + 16 * j + col;
    pv[j] = p < P;
    bj[j] = 0; ohj[j] = 0; owj[j] = 0;
    if (pv[j]) {
      bj[j] = (int)(p / ((long)Ho * Wo));
      int r = (int)(p % ((long)Ho * Wo));
      ohj[j] = r / Wo; owj[j] = r % Wo;
    }
  }

  v8f acc0 = {}, acc1 = {}, acc2 = {}, acc3 = {};
  const _Float16* aw = &sW[(wave * 16 + col) * Ci];

  for (int tap = 0; tap < 9; ++tap) {
    // Per-tile input row base + in-bounds flag for this tap.
    bool inb[4]; size_t bxo[4];
#pragma unroll
    for (int j = 0; j < 4; ++j) {
      int ih = ohj[j] + tap / 3 - pad;
      int iw = owj[j] + tap % 3 - pad;
      inb[j] = pv[j] && ih >= 0 && ih < H && iw >= 0 && iw < W;
      long rowi = ((long)bj[j] * H + ih) * W + iw;
      bxo[j] = (size_t)rowi * (size_t)Ci + (size_t)(kh * 16);
    }
    const _Float16* wp = Wt + ((size_t)tap * Co + coBlock) * (size_t)Ci;

    __syncthreads();                       // previous tap's sW reads complete
#if HAVE_TDM
    if (wave == 0) {
      tdm_load_w_tile((unsigned)(size_t)(&sW[0]), wp, Ci, 128);
      __builtin_amdgcn_s_wait_tensorcnt(0);
    }
#else
    for (int i = threadIdx.x; i < 128 * Ci; i += 256) sW[i] = wp[i];
#endif
    __syncthreads();

#pragma unroll 2
    for (int kc = 0; kc < Ci; kc += 32) {
      // A fragment: lane<16 -> row=col, K = kc+[0..15]
      //             lane>=16 -> row=col, K = kc+[16..31]
      union { v16h v; v8h h[2]; } af;
      af.h[0] = *(const v8h*)&aw[kc + kh * 16];
      af.h[1] = *(const v8h*)&aw[kc + kh * 16 + 8];

      // B fragments: lane -> column col, K = kc + kh*16 + [0..15] contiguous
      v16h b0 = {}, b1 = {}, b2 = {}, b3 = {};
      if (inb[0]) {
        b0 = *(const v16h*)(X + bxo[0] + kc);
        if (kc + 32 < Ci)
          __builtin_prefetch((const void*)(X + bxo[0] + kc + 32), 0, 1);
      }
      if (inb[1]) b1 = *(const v16h*)(X + bxo[1] + kc);
      if (inb[2]) b2 = *(const v16h*)(X + bxo[2] + kc);
      if (inb[3]) b3 = *(const v16h*)(X + bxo[3] + kc);

      acc0 = __builtin_amdgcn_wmma_f32_16x16x32_f16(false, af.v, false, b0,
                                                    (short)0, acc0, false, false);
      acc1 = __builtin_amdgcn_wmma_f32_16x16x32_f16(false, af.v, false, b1,
                                                    (short)0, acc1, false, false);
      acc2 = __builtin_amdgcn_wmma_f32_16x16x32_f16(false, af.v, false, b2,
                                                    (short)0, acc2, false, false);
      acc3 = __builtin_amdgcn_wmma_f32_16x16x32_f16(false, af.v, false, b3,
                                                    (short)0, acc3, false, false);
    }
  }

  // D layout: VGPR j, lanes 0-15 -> M=j ; lanes 16-31 -> M=8+j ; N=lane%16
  const int m0 = coBlock + wave * 16 + kh * 8;
  float bsv[8];
#pragma unroll
  for (int j = 0; j < 8; ++j) bsv[j] = bias[m0 + j];
  const v8f* accs[4] = {&acc0, &acc1, &acc2, &acc3};
#pragma unroll
  for (int t = 0; t < 4; ++t) {
    if (!pv[t]) continue;
    long p = p0 + 16 * t + col;
    float* yp = Y + (size_t)p * Co + m0;
#pragma unroll
    for (int j = 0; j < 8; ++j) yp[j] = (*accs[t])[j] + bsv[j];
  }
}

// --------------------------- support kernels -------------------------------
__global__ void zero_f32_k(float* p, int n) {
  int i = blockIdx.x * 256 + threadIdx.x;
  if (i < n) p[i] = 0.f;
}

// w [Co,Ci,3,3] f32 -> wt [9,Co,Cieff] f16 (zero-pad ci >= Ci)
__global__ void prep_w_k(const float* __restrict__ w, _Float16* __restrict__ wt,
                         int Co, int Ci, int Cieff) {
  int idx = blockIdx.x * 256 + threadIdx.x;
  int total = 9 * Co * Cieff;
  if (idx >= total) return;
  int ci = idx % Cieff;
  int co = (idx / Cieff) % Co;
  int tap = idx / (Cieff * Co);
  float v = (ci < Ci) ? w[((size_t)co * Ci + ci) * 9 + tap] : 0.f;
  wt[idx] = (_Float16)v;
}

// x [4,3,32,32,64] f32 -> xs [256(n*64+t),32,32,32ch] f16 (ch 3..31 zero)
__global__ void prep_x_snn_k(const float* __restrict__ x, _Float16* __restrict__ xs) {
  int idx = blockIdx.x * 256 + threadIdx.x;
  if (idx >= 256 * 32 * 32 * 32) return;
  int cp = idx & 31;
  int w = (idx >> 5) & 31;
  int h = (idx >> 10) & 31;
  int bt = idx >> 15;
  int n = bt >> 6, t = bt & 63;
  float v = 0.f;
  if (cp < 3) v = x[((((size_t)n * 3 + cp) * 32 + h) * 32 + w) * 64 + t];
  xs[idx] = (_Float16)v;
}

// x_ = x.sum(t)/64 -> xa [4,32,32,32ch] f16
__global__ void prep_x_ann_k(const float* __restrict__ x, _Float16* __restrict__ xa) {
  int idx = blockIdx.x * 256 + threadIdx.x;
  if (idx >= 4 * 32 * 32 * 32) return;
  int cp = idx & 31;
  int w = (idx >> 5) & 31;
  int h = (idx >> 10) & 31;
  int n = idx >> 15;
  float v = 0.f;
  if (cp < 3) {
    const float* xb = x + ((((size_t)n * 3 + cp) * 32 + h) * 32 + w) * 64;
    float s = 0.f;
#pragma unroll 8
    for (int t = 0; t < 64; ++t) s += xb[t];
    v = s * 0.015625f;
  }
  xa[idx] = (_Float16)v;
}

// Temporal IF-neuron scan: cin [4*64,H,W,C] f32 -> spikes f16 + per-ch sums
__global__ void scan_tt_k(const float* __restrict__ cin, const float* __restrict__ th,
                          _Float16* __restrict__ sout, float* __restrict__ ssum,
                          int H, int W, int C) {
  int idx = blockIdx.x * 256 + threadIdx.x;
  int total = 4 * H * W * C;
  if (idx >= total) return;
  int c = idx % C;
  int t1 = idx / C;
  int w = t1 % W; t1 /= W;
  int h = t1 % H;
  int n = t1 / H;
  const float thv = th[c];
  const size_t stride = (size_t)H * W * C;
  const size_t b0 = (size_t)n * 64 * stride + (size_t)((h * W + w) * C + c);
  float v = 0.f, acc = 0.f;
  for (int t = 0; t < 64; ++t) {
    v += cin[b0 + (size_t)t * stride];
    float s = (v >= thv) ? 1.f : 0.f;
    v -= s * thv;
    sout[b0 + (size_t)t * stride] = (_Float16)s;
    acc += s;
  }
  atomicAdd(&ssum[c], acc);
}

// y = clip(floor(y*64)/64, 0, 1) -> f16, plus per-channel sums
__global__ void clip_k(const float* __restrict__ ya, _Float16* __restrict__ yo,
                       float* __restrict__ ysum, int NP, int C) {
  int idx = blockIdx.x * 256 + threadIdx.x;
  if (idx >= NP * C) return;
  int c = idx % C;
  float v = ya[idx];
  v = floorf(v * 64.f) * 0.015625f;
  v = v < 0.f ? 0.f : (v > 1.f ? 1.f : v);
  yo[idx] = (_Float16)v;
  atomicAdd(&ysum[c], v);
}

// 2x2 average pool, f16 NHWC in -> f32 NHWC out
__global__ void pool2_k(const _Float16* __restrict__ in, float* __restrict__ outp,
                        int B, int H, int W, int C) {
  int Ho = H / 2, Wo = W / 2;
  long total = (long)B * Ho * Wo * C;
  long idx = (long)blockIdx.x * 256 + threadIdx.x;
  if (idx >= total) return;
  int c = (int)(idx % C);
  long t = idx / C;
  int w2 = (int)(t % Wo); t /= Wo;
  int h2 = (int)(t % Ho);
  long b = t / Ho;
  size_t bi = (((size_t)b * H + 2 * h2) * W + 2 * w2) * C + c;
  size_t rs = (size_t)W * C;
  float s = (float)in[bi] + (float)in[bi + C] + (float)in[bi + rs] + (float)in[bi + rs + C];
  outp[idx] = 0.25f * s;
}

__global__ void thresh_k(const float* __restrict__ thp, const float* __restrict__ ysum,
                         const float* __restrict__ ssum, float* __restrict__ tout,
                         float invmul, int C) {
  int c = blockIdx.x * 256 + threadIdx.x;
  if (c >= C) return;
  float diff = (ysum[c] * 64.f - ssum[c]) * invmul;
  tout[c] = thp[c] - 0.1f * diff;
}

// spikes [256,1,1,1024] -> rate [4,1024] (sum over t)
__global__ void rate_k(const _Float16* __restrict__ xs, float* __restrict__ rate) {
  int idx = blockIdx.x * 256 + threadIdx.x;
  if (idx >= 4096) return;
  int n = idx >> 10, c = idx & 1023;
  float s = 0.f;
  for (int t = 0; t < 64; ++t) s += (float)xs[(((size_t)n * 64 + t) << 10) + c];
  rate[idx] = s;
}

__global__ void fc_k(const float* __restrict__ rate, const float* __restrict__ wc,
                     const float* __restrict__ bc, float* __restrict__ outp) {
  int idx = blockIdx.x * 256 + threadIdx.x;
  if (idx >= 400) return;
  int n = idx / 100, o = idx % 100;
  float s = 0.f;
  for (int c = 0; c < 1024; ++c) s += wc[o * 1024 + c] * rate[n * 1024 + c];
  outp[idx] = s * 0.015625f + bc[o];
}

// ---------------------------------------------------------------------------
extern "C" void kernel_launch(void* const* d_in, const int* in_sizes, int n_in,
                              void* d_out, int out_size, void* d_ws, size_t ws_size,
                              hipStream_t stream) {
  (void)in_sizes; (void)n_in; (void)out_size; (void)ws_size;
  const float *Ws[7], *Bs[7], *Wa[7], *Ba[7];
  for (int l = 0; l < 7; ++l) {
    Ws[l] = (const float*)d_in[4 * l + 0];
    Bs[l] = (const float*)d_in[4 * l + 1];
    Wa[l] = (const float*)d_in[4 * l + 2];
    Ba[l] = (const float*)d_in[4 * l + 3];
  }
  const float* wc = (const float*)d_in[28];
  const float* bc = (const float*)d_in[29];
  const float* x  = (const float*)d_in[30];
  const float* thp[7];
  for (int i = 0; i < 7; ++i) thp[i] = (const float*)d_in[31 + i];
  const float* pp[4];
  for (int i = 0; i < 4; ++i) pp[i] = (const float*)d_in[38 + i];
  float* out = (float*)d_out;

  char* base = (char*)d_ws;
  const size_t MB = 1ull << 20;
  _Float16* WT = (_Float16*)(base);
  _Float16* XS[2] = {(_Float16*)(base + 40 * MB), (_Float16*)(base + 120 * MB)};
  _Float16* XA[2] = {(_Float16*)(base + 200 * MB), (_Float16*)(base + 204 * MB)};
  float* CS  = (float*)(base + 208 * MB);
  float* CA  = (float*)(base + 352 * MB);
  float* ACC = (float*)(base + 356 * MB);

  static const int CoL[7]  = {128, 128, 256, 256, 512, 512, 1024};
  static const int CiL[7]  = {3, 128, 128, 256, 256, 512, 512};
  static const int CieL[7] = {32, 128, 128, 256, 256, 512, 512};
  size_t woff[14]; size_t wa = 0;
  for (int l = 0; l < 7; ++l) {
    size_t sz = (size_t)9 * CoL[l] * CieL[l];
    woff[2 * l] = wa; wa += sz;
    woff[2 * l + 1] = wa; wa += sz;
  }

  // zero accumulators (11 stages x (ssum,ysum) + rate)
  const int accN = 11 * 2048 + 4096;
  zero_f32_k<<<(accN + 255) / 256, 256, 0, stream>>>(ACC, accN);

  // weight transpose/convert
  for (int l = 0; l < 7; ++l) {
    int n = 9 * CoL[l] * CieL[l];
    prep_w_k<<<(n + 255) / 256, 256, 0, stream>>>(Ws[l], WT + woff[2 * l], CoL[l], CiL[l], CieL[l]);
    prep_w_k<<<(n + 255) / 256, 256, 0, stream>>>(Wa[l], WT + woff[2 * l + 1], CoL[l], CiL[l], CieL[l]);
  }
  { int n = 256 * 32 * 32 * 32; prep_x_snn_k<<<(n + 255) / 256, 256, 0, stream>>>(x, XS[0]); }
  { int n = 4 * 32 * 32 * 32;   prep_x_ann_k<<<(n + 255) / 256, 256, 0, stream>>>(x, XA[0]); }

  int sp = 0, ap = 0, stage = 0;

  auto convL = [&](const _Float16* Xin, const _Float16* Wt, const float* bias,
                   float* Y, int Bn, int Hn, int Ci, int Co, int pad) {
    int Ho = pad ? Hn : Hn - 2;
    long P = (long)Bn * Ho * Ho;
    dim3 grid((unsigned)((P + 63) / 64), (unsigned)(Co / 128));
    size_t shmem = (size_t)128 * Ci * sizeof(_Float16);  // per-tap weight panel
    conv3x3_wmma_f16<<<grid, 256, shmem, stream>>>(Xin, Wt, bias, Y, Bn, Hn, Hn, Ci, Co, pad);
  };
  auto tt_after = [&](int Ho, int Co, const float* th, int outOff, float mul) {
    float* ssum = ACC + stage * 2048;
    float* ysum = ssum + 1024;
    { int n = 4 * Ho * Ho * Co;
      scan_tt_k<<<(n + 255) / 256, 256, 0, stream>>>(CS, th, XS[1 - sp], ssum, Ho, Ho, Co); }
    { int n = 4 * Ho * Ho * Co;
      clip_k<<<(n + 255) / 256, 256, 0, stream>>>(CA, XA[1 - ap], ysum, 4 * Ho * Ho, Co); }
    thresh_k<<<(Co + 255) / 256, 256, 0, stream>>>(th, ysum, ssum, out + outOff, 1.0f / mul, Co);
    sp = 1 - sp; ap = 1 - ap; ++stage;
  };
  auto conv_stage = [&](int l, int Hin, int pad, const float* th, int outOff, float mul) {
    convL(XS[sp], WT + woff[2 * l],     Bs[l], CS, 256, Hin, CieL[l], CoL[l], pad);
    convL(XA[ap], WT + woff[2 * l + 1], Ba[l], CA, 4,   Hin, CieL[l], CoL[l], pad);
    tt_after(pad ? Hin : Hin - 2, CoL[l], th, outOff, mul);
  };
  auto pool_stage = [&](int C, int Hin, const float* th, int outOff, float mul) {
    { long n = (long)256 * (Hin / 2) * (Hin / 2) * C;
      pool2_k<<<(unsigned)((n + 255) / 256), 256, 0, stream>>>(XS[sp], CS, 256, Hin, Hin, C); }
    { long n = (long)4 * (Hin / 2) * (Hin / 2) * C;
      pool2_k<<<(unsigned)((n + 255) / 256), 256, 0, stream>>>(XA[ap], CA, 4, Hin, Hin, C); }
    tt_after(Hin / 2, C, th, outOff, mul);
  };

  conv_stage(0, 32, 1, thp[0], 400, 262144.f);   // t1
  conv_stage(1, 32, 1, thp[1], 528, 262144.f);   // t2
  pool_stage(128, 32, pp[0], 3216, 65536.f);     // q1
  conv_stage(2, 16, 1, thp[2], 656, 65536.f);    // t3
  conv_stage(3, 16, 1, thp[3], 912, 65536.f);    // t4
  pool_stage(256, 16, pp[1], 3344, 16384.f);     // q2
  conv_stage(4, 8, 1, thp[4], 1168, 16384.f);    // t5
  conv_stage(5, 8, 1, thp[5], 1680, 16384.f);    // t6
  pool_stage(512, 8, pp[2], 3600, 4096.f);       // q3
  conv_stage(6, 4, 0, thp[6], 2192, 1024.f);     // t7 (pad=0: 4x4 -> 2x2)
  pool_stage(1024, 2, pp[3], 4112, 256.f);       // q4 (2x2 -> 1x1)

  float* rate = ACC + 11 * 2048;
  rate_k<<<16, 256, 0, stream>>>(XS[sp], rate);
  fc_k<<<2, 256, 0, stream>>>(rate, wc, bc, out);
}
